// RetICLBase_49125835931791
// MI455X (gfx1250) — compile-verified, matrix operand order checked by reading, not run on hardware
//
#include <hip/hip_runtime.h>
#include <hip/hip_bf16.h>

// ---------------------------------------------------------------------------
// RetICL activations on MI455X (gfx1250, wave32, WMMA).
// Big GEMM: BF16x3 split-precision (hi*hi + hi*lo + lo*hi) with
// v_wmma_f32_16x16x32_bf16, 2x2 register blocking per wave, enc strip staged
// in LDS per block via GLOBAL_LOAD_ASYNC_TO_LDS_B128 (ASYNCcnt tracked).
// Small q GEMM: exact fp32 v_wmma_f32_16x16x4_f32.
// ---------------------------------------------------------------------------

typedef __attribute__((ext_vector_type(16))) __bf16 v16bf;
typedef __attribute__((ext_vector_type(8)))  float  v8f;
typedef __attribute__((ext_vector_type(2)))  float  v2f;

#define BB 128
#define SS 8
#define HH 768
#define EE 768
#define NN 50000
#define NPAD 50016           // padded to 3126 n-tiles (even # of 16-row tiles)
#define MROWS (BB * SS)      // 1024
#define OUTC  (NN + 1)       // 50001
#define NSTRIP 32            // enc rows staged in LDS per block (2 n-tiles)
#define PLANE_BYTES (NSTRIP * EE * 2)   // 49152 bytes per hi/lo plane

__device__ __forceinline__ unsigned short f2bf_rne(float x) {
    unsigned u = __float_as_uint(x);
    return (unsigned short)((u + 0x7FFFu + ((u >> 16) & 1u)) >> 16);
}
__device__ __forceinline__ float bf2f(unsigned short h) {
    return __uint_as_float(((unsigned)h) << 16);
}

union FragBF {
    uint4 u[2];
    v16bf v;
};

// Load one bf16 16x32 A/B fragment: two K-contiguous 16B chunks per lane.
__device__ __forceinline__ v16bf load_frag(const unsigned short* p) {
    FragBF f;
    f.u[0] = *(const uint4*)(p);
    f.u[1] = *(const uint4*)(p + 16);
    return f.v;
}

// ---------------------------------------------------------------------------
// Kernel 1: split enc (N x E fp32) into hi/lo bf16 planes, padded to NPAD
// rows (pad rows zero-filled).  4 elements/thread;
// NPAD*EE = 38,412,288 = 37512 * 256 * 4 exactly.
// ---------------------------------------------------------------------------
__global__ void enc_split_kernel(const float* __restrict__ enc,
                                 unsigned short* __restrict__ ehi,
                                 unsigned short* __restrict__ elo) {
    size_t i = ((size_t)blockIdx.x * blockDim.x + threadIdx.x) * 4;
    if (i >= (size_t)NPAD * EE) return;
    float4 x = make_float4(0.f, 0.f, 0.f, 0.f);
    if (i < (size_t)NN * EE) x = *(const float4*)(enc + i);
    unsigned short h0 = f2bf_rne(x.x), h1 = f2bf_rne(x.y);
    unsigned short h2 = f2bf_rne(x.z), h3 = f2bf_rne(x.w);
    unsigned short l0 = f2bf_rne(x.x - bf2f(h0));
    unsigned short l1 = f2bf_rne(x.y - bf2f(h1));
    unsigned short l2 = f2bf_rne(x.z - bf2f(h2));
    unsigned short l3 = f2bf_rne(x.w - bf2f(h3));
    uint2 hp, lp;
    hp.x = (unsigned)h0 | ((unsigned)h1 << 16);
    hp.y = (unsigned)h2 | ((unsigned)h3 << 16);
    lp.x = (unsigned)l0 | ((unsigned)l1 << 16);
    lp.y = (unsigned)l2 | ((unsigned)l3 << 16);
    *(uint2*)(ehi + i) = hp;
    *(uint2*)(elo + i) = lp;
}

// ---------------------------------------------------------------------------
// Kernel 2: q = ls @ bilinear (1024 x 768) in exact fp32 WMMA 16x16x4.
// Epilogue splits q into hi/lo bf16 (row-major, K-major for the next GEMM).
// One 16x16 tile per wave; 3072 tiles = 384 blocks * 8 waves.
// ---------------------------------------------------------------------------
__global__ void q_gemm_kernel(const float* __restrict__ latent,
                              const float* __restrict__ bilinear,
                              unsigned short* __restrict__ qhi,
                              unsigned short* __restrict__ qlo) {
    const int lane = threadIdx.x & 31;
    const int wave = threadIdx.x >> 5;
    const int tile = blockIdx.x * 8 + wave;    // [0, 3072)
    const int mt = tile & 63;                  // 64 m-tiles
    const int et = tile >> 6;                  // 48 e-tiles
    const int mbase = mt * 16, ebase = et * 16;
    const int hi16 = lane >> 4;                // 0/1
    const int l16  = lane & 15;

    const int m = mbase + l16;
    const int b = m >> 3, s = m & 7;
    const float* __restrict__ arow = latent + (size_t)(b * (SS + 1) + s) * HH;
    const int e = ebase + l16;

    v8f c = {};
#pragma unroll 4
    for (int kb = 0; kb < HH; kb += 4) {
        const int k0 = kb + hi16 * 2;
        v2f a;
        a.x = arow[k0];
        a.y = arow[k0 + 1];
        v2f bf;
        bf.x = bilinear[(size_t)k0 * EE + e];
        bf.y = bilinear[(size_t)(k0 + 1) * EE + e];
        c = __builtin_amdgcn_wmma_f32_16x16x4_f32(
                false, a, false, bf, (short)0, c, false, false);
    }
#pragma unroll
    for (int v = 0; v < 8; ++v) {
        const int mm = mbase + v + hi16 * 8;
        const float x = c[v];
        const unsigned short h = f2bf_rne(x);
        const unsigned short l = f2bf_rne(x - bf2f(h));
        qhi[(size_t)mm * EE + e] = h;
        qlo[(size_t)mm * EE + e] = l;
    }
}

// ---------------------------------------------------------------------------
// Kernel 3: act^T = enc x q^T + bias, BF16x3 split WMMA, fp32 accumulate.
// Block = 8 waves; block stages a 32-row enc strip (hi+lo, 96 KB) into
// dynamic LDS with async global->LDS b128 copies, then each wave computes a
// 2(n) x 2(m) tile block: 12 v_wmma per K=32 step vs 8 fragment loads
// (4 ds_load + 4 global_load).  Grid: 1563 n-strips * 4 m-blocks = 6252.
// ---------------------------------------------------------------------------
__global__ void act_gemm_kernel(const unsigned short* __restrict__ ehi,
                                const unsigned short* __restrict__ elo,
                                const unsigned short* __restrict__ qhi,
                                const unsigned short* __restrict__ qlo,
                                const float* __restrict__ bias,
                                float* __restrict__ out) {
    extern __shared__ unsigned short smem[];     // 96 KB dynamic LDS
    unsigned short* sAh = smem;                  // [NSTRIP][EE] hi plane
    unsigned short* sAl = smem + NSTRIP * EE;    // [NSTRIP][EE] lo plane

    const int tid  = threadIdx.x;
    const int lane = tid & 31;
    const int wave = tid >> 5;
    const int ntp  = blockIdx.x >> 2;            // [0, 1563) n-strip
    const int mb   = blockIdx.x & 3;
    const int mtp  = (mb << 3) + wave;           // [0, 32) m-pair
    const int nbase0 = ntp * NSTRIP;             // first of 2 n-tiles
    const int nbase1 = nbase0 + 16;
    const int mbase0 = mtp * 32;                 // first of 2 m-tiles
    const int mbase1 = mbase0 + 16;

    // ---- stage enc strip (48 KB/plane, contiguous) via async LDS DMA ----
    // Dynamic LDS starts at groupstaticsize (no static LDS in this kernel),
    // so the VDST operand (raw LDS byte offset) is base + linear offset.
    {
        const char* gh = (const char*)(ehi + (size_t)nbase0 * EE);
        const char* gl = (const char*)(elo + (size_t)nbase0 * EE);
        const unsigned ldsbase = __builtin_amdgcn_groupstaticsize();
#pragma unroll
        for (int i = 0; i < (PLANE_BYTES / 16) / 256; ++i) {   // 12 chunks
            const unsigned off = (unsigned)(i * 4096 + tid * 16);
            asm volatile("global_load_async_to_lds_b128 %0, %1, off"
                         :: "v"(ldsbase + off), "v"(gh + off)
                         : "memory");
            asm volatile("global_load_async_to_lds_b128 %0, %1, off"
                         :: "v"(ldsbase + PLANE_BYTES + off), "v"(gl + off)
                         : "memory");
        }
        asm volatile("s_wait_asynccnt 0" ::: "memory");
    }
    __syncthreads();

    const int l16   = lane & 15;
    const int khalf = (lane >> 4) << 3;          // 0 or 8

    const unsigned short* sa0h = sAh + (size_t)l16 * EE;
    const unsigned short* sa1h = sAh + (size_t)(16 + l16) * EE;
    const unsigned short* sa0l = sAl + (size_t)l16 * EE;
    const unsigned short* sa1l = sAl + (size_t)(16 + l16) * EE;
    const unsigned short* gb0h = qhi + (size_t)(mbase0 + l16) * EE;
    const unsigned short* gb1h = qhi + (size_t)(mbase1 + l16) * EE;
    const unsigned short* gb0l = qlo + (size_t)(mbase0 + l16) * EE;
    const unsigned short* gb1l = qlo + (size_t)(mbase1 + l16) * EE;

    const float bv = bias[0];
    v8f c00 = { bv, bv, bv, bv, bv, bv, bv, bv };   // (n0, m0)
    v8f c01 = c00;                                  // (n0, m1)
    v8f c10 = c00;                                  // (n1, m0)
    v8f c11 = c00;                                  // (n1, m1)

#pragma unroll 2
    for (int kb = 0; kb < EE; kb += 32) {
        const int ko = kb + khalf;
        const v16bf a0h = load_frag(sa0h + ko);     // ds_load_b128 x2
        const v16bf a0l = load_frag(sa0l + ko);
        const v16bf a1h = load_frag(sa1h + ko);
        const v16bf a1l = load_frag(sa1l + ko);
        const v16bf b0h = load_frag(gb0h + ko);     // global_load_b128 x2
        const v16bf b0l = load_frag(gb0l + ko);
        const v16bf b1h = load_frag(gb1h + ko);
        const v16bf b1l = load_frag(gb1l + ko);

        c00 = __builtin_amdgcn_wmma_f32_16x16x32_bf16(false, a0h, false, b0h, (short)0, c00, false, false);
        c00 = __builtin_amdgcn_wmma_f32_16x16x32_bf16(false, a0h, false, b0l, (short)0, c00, false, false);
        c00 = __builtin_amdgcn_wmma_f32_16x16x32_bf16(false, a0l, false, b0h, (short)0, c00, false, false);

        c01 = __builtin_amdgcn_wmma_f32_16x16x32_bf16(false, a0h, false, b1h, (short)0, c01, false, false);
        c01 = __builtin_amdgcn_wmma_f32_16x16x32_bf16(false, a0h, false, b1l, (short)0, c01, false, false);
        c01 = __builtin_amdgcn_wmma_f32_16x16x32_bf16(false, a0l, false, b1h, (short)0, c01, false, false);

        c10 = __builtin_amdgcn_wmma_f32_16x16x32_bf16(false, a1h, false, b0h, (short)0, c10, false, false);
        c10 = __builtin_amdgcn_wmma_f32_16x16x32_bf16(false, a1h, false, b0l, (short)0, c10, false, false);
        c10 = __builtin_amdgcn_wmma_f32_16x16x32_bf16(false, a1l, false, b0h, (short)0, c10, false, false);

        c11 = __builtin_amdgcn_wmma_f32_16x16x32_bf16(false, a1h, false, b1h, (short)0, c11, false, false);
        c11 = __builtin_amdgcn_wmma_f32_16x16x32_bf16(false, a1h, false, b1l, (short)0, c11, false, false);
        c11 = __builtin_amdgcn_wmma_f32_16x16x32_bf16(false, a1l, false, b1h, (short)0, c11, false, false);
    }

    // ---- store: D lane col m = l16, VGPR v -> n = nbase + khalf + v ----
    {
        float* o = out + (size_t)(mbase0 + l16) * OUTC + (nbase0 + khalf);
#pragma unroll
        for (int v = 0; v < 8; ++v) o[v] = c00[v];
        o = out + (size_t)(mbase1 + l16) * OUTC + (nbase0 + khalf);
#pragma unroll
        for (int v = 0; v < 8; ++v) o[v] = c01[v];
    }
    if (nbase1 < NN) {   // uniform: skips only the padded tile (ntp == 1562)
        float* o = out + (size_t)(mbase0 + l16) * OUTC + (nbase1 + khalf);
#pragma unroll
        for (int v = 0; v < 8; ++v) o[v] = c10[v];
        o = out + (size_t)(mbase1 + l16) * OUTC + (nbase1 + khalf);
#pragma unroll
        for (int v = 0; v < 8; ++v) o[v] = c11[v];
    }
}

// ---------------------------------------------------------------------------
// Kernel 4: causal -inf scatter, EOS column, value estimates.
// One thread per (b, t) row; 1024 threads total.
// ---------------------------------------------------------------------------
__global__ void mask_value_kernel(const float* __restrict__ latent,
                                  const float* __restrict__ vfe_w,
                                  const float* __restrict__ vfe_b,
                                  const int* __restrict__ pei,
                                  const int* __restrict__ seq_len,
                                  float* __restrict__ out) {
    const int i = blockIdx.x * blockDim.x + threadIdx.x;
    if (i >= MROWS) return;
    const int b = i >> 3, t = i & 7;
    float* __restrict__ row = out + (size_t)i * OUTC;
    const float NEGINF = -__builtin_huge_valf();

    row[NN] = NEGINF;                        // EOS column, always -inf
    if (t < seq_len[b]) {
        for (int u = 0; u < t; ++u) row[pei[b * SS + u]] = NEGINF;
    }

    const float4* __restrict__ lr = (const float4*)(latent + (size_t)(b * (SS + 1) + t) * HH);
    const float4* __restrict__ wv = (const float4*)vfe_w;
    float acc = 0.f;
#pragma unroll 8
    for (int kk = 0; kk < HH / 4; ++kk) {
        const float4 a = lr[kk], w = wv[kk];
        acc += a.x * w.x + a.y * w.y + a.z * w.z + a.w * w.w;
    }
    out[(size_t)MROWS * OUTC + i] = acc + vfe_b[0];
}

// ---------------------------------------------------------------------------
// Launcher.  Workspace layout (needs ~157 MB of d_ws):
//   [0)        q_hi   bf16 1024x768          (1.5 MB)
//   [1.5 MB)   q_lo   bf16 1024x768          (1.5 MB)
//   [3 MB)     enc_hi bf16 50016x768 padded  (76.8 MB)
//   [79.9 MB)  enc_lo bf16 50016x768 padded  (76.8 MB)
// ---------------------------------------------------------------------------
extern "C" void kernel_launch(void* const* d_in, const int* in_sizes, int n_in,
                              void* d_out, int out_size, void* d_ws, size_t ws_size,
                              hipStream_t stream) {
    (void)in_sizes; (void)n_in; (void)out_size; (void)ws_size;
    const float* latent = (const float*)d_in[0];
    const float* enc    = (const float*)d_in[1];
    const float* bil    = (const float*)d_in[2];
    const float* bias   = (const float*)d_in[3];
    const float* vfe_w  = (const float*)d_in[4];
    const float* vfe_b  = (const float*)d_in[5];
    const int*   pei    = (const int*)d_in[6];
    const int*   seqlen = (const int*)d_in[7];
    float* out = (float*)d_out;

    unsigned char* ws = (unsigned char*)d_ws;
    unsigned short* qhi = (unsigned short*)(ws);
    unsigned short* qlo = (unsigned short*)(ws + (size_t)MROWS * EE * 2);
    unsigned short* ehi = (unsigned short*)(ws + (size_t)MROWS * EE * 4);
    unsigned short* elo = (unsigned short*)(ws + (size_t)MROWS * EE * 4
                                               + (size_t)NPAD * EE * 2);

    // 1) split enc into bf16 hi/lo, zero-padded to NPAD rows
    enc_split_kernel<<<37512, 256, 0, stream>>>(enc, ehi, elo);
    // 2) q = ls @ bilinear (exact fp32 WMMA), split epilogue
    q_gemm_kernel<<<384, 256, 0, stream>>>(latent, bil, qhi, qlo);
    // 3) act = q @ enc^T + bias via BF16x3 WMMA, 2x2 blocking, async-LDS A
    act_gemm_kernel<<<6252, 256, 96 * 1024, stream>>>(ehi, elo, qhi, qlo, bias, out);
    // 4) -inf mask, EOS column, value estimates
    mask_value_kernel<<<4, 256, 0, stream>>>(latent, vfe_w, vfe_b, pei, seqlen, out);
}
